// ablation_1_29575144801048
// MI455X (gfx1250) — matrix operand order repaired and checked by reading, not compiled, not run on hardware
//
#include <hip/hip_runtime.h>
#include <hip/hip_bf16.h>

typedef __bf16 bf16_t;
typedef __attribute__((ext_vector_type(16))) __bf16 v16bf;
typedef __attribute__((ext_vector_type(8)))  __bf16 v8bf;
typedef __attribute__((ext_vector_type(8)))  float  v8f;

#define T_STEPS 2
#define N_NODES 50000
#define N_EDGES 400000
#define F_IN    24
#define H       64

#define KK_NE 1    // K=32  (24 valid)
#define KK_EB 14   // K=448
#define KK_NB 10   // K=320
#define KK_D1 2    // K=64

#define WMMA_BF16(A,B,C) \
  __builtin_amdgcn_wmma_f32_16x16x32_bf16(false,(A),false,(B),(short)0,(C),false,false)

static __device__ __forceinline__ v16bf mk_frag(v8bf lo, v8bf hi) {
  v16bf a;
#pragma unroll
  for (int i = 0; i < 8; ++i) { a[i] = lo[i]; a[i + 8] = hi[i]; }
  return a;
}

// ---------------------------------------------------------------------------
// Repack a fp32 [K,64] weight into WMMA B-fragment order (bf16).
// Fragment (kk, c, lane) holds 16 values B[kk*32 + (lane>=16?8:0) + i + (i>=8?8:0)][c*16 + (lane&15)].
// ---------------------------------------------------------------------------
__global__ void repack_weights_kernel(const float* __restrict__ W, bf16_t* __restrict__ Wf,
                                      int KK, int Kvalid) {
  int idx = blockIdx.x * blockDim.x + threadIdx.x;
  int total = KK * 4 * 32;
  if (idx >= total) return;
  int lane = idx & 31;
  int cc   = (idx >> 5) & 3;
  int kk   = idx >> 7;
  int col  = cc * 16 + (lane & 15);
  int kb   = kk * 32 + ((lane >> 4) ? 8 : 0);
#pragma unroll
  for (int i = 0; i < 16; ++i) {
    int k = kb + i + ((i >= 8) ? 8 : 0);
    float v = (k < Kvalid) ? W[(size_t)k * H + col] : 0.f;
    Wf[(size_t)idx * 16 + i] = (bf16_t)v;
  }
}

// ---------------------------------------------------------------------------
// h_g = relu(global_attr[0] * W_ge + b_ge)   (only enc_g[0] is ever used)
// ---------------------------------------------------------------------------
__global__ void encode_global_kernel(const float* __restrict__ ga,
                                     const float* __restrict__ W_ge,
                                     const float* __restrict__ b_ge,
                                     float* __restrict__ h_g) {
  int j = threadIdx.x;
  if (j < H) {
    float v = ga[0] * W_ge[j] + b_ge[j];
    h_g[j] = v > 0.f ? v : 0.f;
  }
}

// ---------------------------------------------------------------------------
// enc_x = relu(node_attr @ W_ne + b_ne)  over T*N rows, K=24 padded to 32.
// ---------------------------------------------------------------------------
__global__ __launch_bounds__(256)
void encode_nodes_kernel(const float* __restrict__ na, const bf16_t* __restrict__ Wf,
                         const float* __restrict__ b_ne, float* __restrict__ enc_x,
                         int rows) {
  __shared__ bf16_t ldsW[KK_NE * 4 * 32 * 16];
  for (int i = threadIdx.x; i < KK_NE * 4 * 32 * 16 / 8; i += blockDim.x)
    ((v8bf*)ldsW)[i] = ((const v8bf*)Wf)[i];
  __syncthreads();

  int lane = threadIdx.x & 31;
  int wave = threadIdx.x >> 5;
  int gw   = blockIdx.x * (blockDim.x >> 5) + wave;
  int nw   = gridDim.x * (blockDim.x >> 5);
  int row  = lane & 15;
  int koff = (lane >> 4) * 8;
  int coln = lane & 15;
  int mb   = (lane >> 4) * 8;
  int ntiles = rows >> 4;

  for (int tile = gw; tile < ntiles; tile += nw) {
    const float* r0 = na + (size_t)(tile * 16 + row) * F_IN;
    v8bf lo, hi;
#pragma unroll
    for (int k = 0; k < 8; ++k) {
      int fa = koff + k;
      int fb = 16 + koff + k;
      lo[k] = (bf16_t)((fa < F_IN) ? r0[fa] : 0.f);
      hi[k] = (bf16_t)((fb < F_IN) ? r0[fb] : 0.f);
    }
    v16bf a = mk_frag(lo, hi);
    v8f acc[4] = {};
#pragma unroll
    for (int c = 0; c < 4; ++c) {
      v16bf b = *(const v16bf*)(ldsW + ((c * 32 + lane) << 4));
      acc[c] = WMMA_BF16(a, b, acc[c]);
    }
#pragma unroll
    for (int c = 0; c < 4; ++c) {
      int n = c * 16 + coln;
      float bias = b_ne[n];
#pragma unroll
      for (int v = 0; v < 8; ++v) {
        float val = acc[c][v] + bias;
        enc_x[(size_t)(tile * 16 + mb + v) * H + n] = val > 0.f ? val : 0.f;
      }
    }
  }
}

// ---------------------------------------------------------------------------
// Per-step prep: cxb = bf16([enc_x[t], h_x]); zero sent/recv; gb = bf16(h_g);
// zero sumX/sumE.
// ---------------------------------------------------------------------------
__global__ void prep_step_kernel(const float* __restrict__ encx_t, const float* __restrict__ hx,
                                 bf16_t* __restrict__ cxb, float* __restrict__ sent,
                                 float* __restrict__ recv, const float* __restrict__ h_g,
                                 bf16_t* __restrict__ gb, float* __restrict__ sumX,
                                 float* __restrict__ sumE, int nN) {
  int i = blockIdx.x * blockDim.x + threadIdx.x;
  if (i < nN) {
    const float4* e4 = (const float4*)(encx_t + (size_t)i * H);
    const float4* h4 = (const float4*)(hx + (size_t)i * H);
    bf16_t* dst = cxb + (size_t)i * 128;
    float4* s4 = (float4*)(sent + (size_t)i * H);
    float4* r4 = (float4*)(recv + (size_t)i * H);
    float4 z = {0.f, 0.f, 0.f, 0.f};
#pragma unroll
    for (int q = 0; q < 16; ++q) {
      float4 a = e4[q], b = h4[q];
      dst[4*q+0] = (bf16_t)a.x; dst[4*q+1] = (bf16_t)a.y;
      dst[4*q+2] = (bf16_t)a.z; dst[4*q+3] = (bf16_t)a.w;
      dst[64+4*q+0] = (bf16_t)b.x; dst[64+4*q+1] = (bf16_t)b.y;
      dst[64+4*q+2] = (bf16_t)b.z; dst[64+4*q+3] = (bf16_t)b.w;
      s4[q] = z; r4[q] = z;
    }
  }
  if (blockIdx.x == 0 && threadIdx.x < H) {
    gb[threadIdx.x] = (bf16_t)h_g[threadIdx.x];
    sumX[threadIdx.x] = 0.f;
    sumE[threadIdx.x] = 0.f;
  }
}

// ---------------------------------------------------------------------------
// EdgeBlock A-chunk: 8 contiguous features starting at f (f multiple of 8).
// Feature map: [0,64) enc_e inline, [64,128) h_e, [128,256) cxb[sender],
// [256,384) cxb[receiver], [384,448) gb.
// ---------------------------------------------------------------------------
static __device__ __forceinline__ v8bf edge_chunk(
    int f, int e, int s, int r, float ea,
    const float* __restrict__ W_ee, const float* __restrict__ b_ee,
    const float* __restrict__ new_e,
    const bf16_t* __restrict__ cxb, const bf16_t* __restrict__ gb,
    int first_step) {
  v8bf o;
  if (f < 128) {
    if (f < 64 || first_step) {
      int j0 = f & 63;
#pragma unroll
      for (int k = 0; k < 8; ++k) {
        float v = ea * W_ee[j0 + k] + b_ee[j0 + k];
        o[k] = (bf16_t)(v > 0.f ? v : 0.f);
      }
    } else {
      const float* src = new_e + (size_t)e * H + (f - 64);
#pragma unroll
      for (int k = 0; k < 8; ++k) o[k] = (bf16_t)src[k];
    }
    return o;
  }
  if (f < 256) return *(const v8bf*)(cxb + (size_t)s * 128 + (f - 128));
  if (f < 384) return *(const v8bf*)(cxb + (size_t)r * 128 + (f - 256));
  return *(const v8bf*)(gb + (f - 384));
}

// Two 16-edge M-tiles per wave: each B fragment feeds two WMMAs, halving the
// LDS weight-stream per edge. A-chunk gathers are software-pipelined one
// k-step ahead of the WMMAs.
__global__ __launch_bounds__(256)
void edge_block_kernel(const float* __restrict__ ea_t,
                       const float* __restrict__ W_ee, const float* __restrict__ b_ee,
                       const bf16_t* __restrict__ cxb,
                       float* __restrict__ new_e,
                       const int* __restrict__ senders, const int* __restrict__ receivers,
                       const bf16_t* __restrict__ Wf, const float* __restrict__ b_eb,
                       const bf16_t* __restrict__ gb,
                       float* __restrict__ sent, float* __restrict__ recv,
                       float* __restrict__ sumE,
                       int nE, int first_step) {
  __shared__ bf16_t ldsW[KK_EB * 4 * 32 * 16];
  __shared__ float  ldsSum[H];
  for (int i = threadIdx.x; i < KK_EB * 4 * 32 * 16 / 8; i += blockDim.x)
    ((v8bf*)ldsW)[i] = ((const v8bf*)Wf)[i];
  if (threadIdx.x < H) ldsSum[threadIdx.x] = 0.f;
  __syncthreads();

  int lane = threadIdx.x & 31;
  int wave = threadIdx.x >> 5;
  int gw   = blockIdx.x * (blockDim.x >> 5) + wave;
  int nw   = gridDim.x * (blockDim.x >> 5);
  int row  = lane & 15;
  int koff = (lane >> 4) * 8;
  int coln = lane & 15;
  int mb   = (lane >> 4) * 8;
  int ntp  = nE >> 5;   // pairs of 16-edge tiles

  for (int tp = gw; tp < ntp; tp += nw) {
    int e0 = tp * 32 + row;
    int e1 = e0 + 16;
    int s0 = senders[e0],   s1 = senders[e1];
    int r0 = receivers[e0], r1 = receivers[e1];
    float ea0 = ea_t[e0], ea1 = ea_t[e1];

    v8f acc[2][4] = {};

    // prologue: gather k-step 0
    v8bf lo0 = edge_chunk(koff,      e0, s0, r0, ea0, W_ee, b_ee, new_e, cxb, gb, first_step);
    v8bf hi0 = edge_chunk(koff + 16, e0, s0, r0, ea0, W_ee, b_ee, new_e, cxb, gb, first_step);
    v8bf lo1 = edge_chunk(koff,      e1, s1, r1, ea1, W_ee, b_ee, new_e, cxb, gb, first_step);
    v8bf hi1 = edge_chunk(koff + 16, e1, s1, r1, ea1, W_ee, b_ee, new_e, cxb, gb, first_step);

#pragma unroll
    for (int kk = 0; kk < KK_EB; ++kk) {
      v16bf a0 = mk_frag(lo0, hi0);
      v16bf a1 = mk_frag(lo1, hi1);
      if (kk + 1 < KK_EB) {               // prefetch next k-step's A chunks
        int fN = (kk + 1) * 32 + koff;
        lo0 = edge_chunk(fN,      e0, s0, r0, ea0, W_ee, b_ee, new_e, cxb, gb, first_step);
        hi0 = edge_chunk(fN + 16, e0, s0, r0, ea0, W_ee, b_ee, new_e, cxb, gb, first_step);
        lo1 = edge_chunk(fN,      e1, s1, r1, ea1, W_ee, b_ee, new_e, cxb, gb, first_step);
        hi1 = edge_chunk(fN + 16, e1, s1, r1, ea1, W_ee, b_ee, new_e, cxb, gb, first_step);
      }
#pragma unroll
      for (int c = 0; c < 4; ++c) {
        v16bf b = *(const v16bf*)(ldsW + ((((kk * 4 + c) * 32) + lane) << 4));
        acc[0][c] = WMMA_BF16(a0, b, acc[0][c]);
        acc[1][c] = WMMA_BF16(a1, b, acc[1][c]);
      }
    }

    // epilogue: bias + relu, store new_e, scatter-add sent/recv, column sums
#pragma unroll
    for (int mt = 0; mt < 2; ++mt) {
      int tbase = tp * 32 + mt * 16;
      int srow[8], rrow[8];
#pragma unroll
      for (int v = 0; v < 8; ++v) {
        srow[v] = senders[tbase + mb + v];
        rrow[v] = receivers[tbase + mb + v];
      }
#pragma unroll
      for (int c = 0; c < 4; ++c) {
        int n = c * 16 + coln;
        float bias = b_eb[n];
        float colsum = 0.f;
#pragma unroll
        for (int v = 0; v < 8; ++v) {
          float val = acc[mt][c][v] + bias;
          val = val > 0.f ? val : 0.f;
          new_e[(size_t)(tbase + mb + v) * H + n] = val;
          atomicAdd(&sent[(size_t)srow[v] * H + n], val);
          atomicAdd(&recv[(size_t)rrow[v] * H + n], val);
          colsum += val;
        }
        atomicAdd(&ldsSum[n], colsum);
      }
    }
  }
  __syncthreads();
  if (threadIdx.x < H) atomicAdd(&sumE[threadIdx.x], ldsSum[threadIdx.x]);
}

// ---------------------------------------------------------------------------
// NodeBlock: nb_in = [cxb (bf16,128) | sent 64 | recv 64 | gb 64], K = 320.
// ---------------------------------------------------------------------------
static __device__ __forceinline__ v8bf node_chunk(
    int f, int i, const bf16_t* __restrict__ cxb,
    const float* __restrict__ sent, const float* __restrict__ recv,
    const bf16_t* __restrict__ gb) {
  if (f < 128) return *(const v8bf*)(cxb + (size_t)i * 128 + f);
  v8bf o;
  if (f < 192) {
    const float* s = sent + (size_t)i * H + (f - 128);
#pragma unroll
    for (int k = 0; k < 8; ++k) o[k] = (bf16_t)s[k];
    return o;
  }
  if (f < 256) {
    const float* s = recv + (size_t)i * H + (f - 192);
#pragma unroll
    for (int k = 0; k < 8; ++k) o[k] = (bf16_t)s[k];
    return o;
  }
  return *(const v8bf*)(gb + (f - 256));
}

__global__ __launch_bounds__(256)
void node_block_kernel(const bf16_t* __restrict__ cxb,
                       const float* __restrict__ sent, const float* __restrict__ recv,
                       const bf16_t* __restrict__ gb,
                       const bf16_t* __restrict__ Wf, const float* __restrict__ b_nb,
                       float* __restrict__ new_x, float* __restrict__ sumX, int nN) {
  __shared__ bf16_t ldsW[KK_NB * 4 * 32 * 16];
  __shared__ float  ldsSum[H];
  for (int i = threadIdx.x; i < KK_NB * 4 * 32 * 16 / 8; i += blockDim.x)
    ((v8bf*)ldsW)[i] = ((const v8bf*)Wf)[i];
  if (threadIdx.x < H) ldsSum[threadIdx.x] = 0.f;
  __syncthreads();

  int lane = threadIdx.x & 31;
  int wave = threadIdx.x >> 5;
  int gw   = blockIdx.x * (blockDim.x >> 5) + wave;
  int nw   = gridDim.x * (blockDim.x >> 5);
  int row  = lane & 15;
  int koff = (lane >> 4) * 8;
  int coln = lane & 15;
  int mb   = (lane >> 4) * 8;
  int ntiles = nN >> 4;

  for (int tile = gw; tile < ntiles; tile += nw) {
    int node = tile * 16 + row;
    v8f acc[4] = {};
#pragma unroll
    for (int kk = 0; kk < KK_NB; ++kk) {
      int fA = kk * 32 + koff;
      v8bf lo = node_chunk(fA,      node, cxb, sent, recv, gb);
      v8bf hi = node_chunk(fA + 16, node, cxb, sent, recv, gb);
      v16bf a = mk_frag(lo, hi);
#pragma unroll
      for (int c = 0; c < 4; ++c) {
        v16bf b = *(const v16bf*)(ldsW + ((((kk * 4 + c) * 32) + lane) << 4));
        acc[c] = WMMA_BF16(a, b, acc[c]);
      }
    }
#pragma unroll
    for (int c = 0; c < 4; ++c) {
      int n = c * 16 + coln;
      float bias = b_nb[n];
      float colsum = 0.f;
#pragma unroll
      for (int v = 0; v < 8; ++v) {
        float val = acc[c][v] + bias;
        val = val > 0.f ? val : 0.f;
        new_x[(size_t)(tile * 16 + mb + v) * H + n] = val;
        colsum += val;
      }
      atomicAdd(&ldsSum[n], colsum);
    }
  }
  __syncthreads();
  if (threadIdx.x < H) atomicAdd(&sumX[threadIdx.x], ldsSum[threadIdx.x]);
}

// ---------------------------------------------------------------------------
// GlobalBlock: h_g = relu([sumX/N, sumE/E, h_g] @ W_gb + b_gb)
// ---------------------------------------------------------------------------
__global__ void global_block_kernel(const float* __restrict__ sumX,
                                    const float* __restrict__ sumE,
                                    float* __restrict__ h_g,
                                    const float* __restrict__ W_gb,
                                    const float* __restrict__ b_gb,
                                    int nN, int nE) {
  __shared__ float gin[192];
  int j = threadIdx.x;
  if (j < H) {
    gin[j]       = sumX[j] / (float)nN;
    gin[H + j]   = sumE[j] / (float)nE;
    gin[128 + j] = h_g[j];
  }
  __syncthreads();
  if (j < H) {
    float v = b_gb[j];
    for (int k = 0; k < 192; ++k) v += gin[k] * W_gb[(size_t)k * H + j];
    h_g[j] = v > 0.f ? v : 0.f;
  }
}

// ---------------------------------------------------------------------------
// Decoder: out = relu(new_x @ W_d1 + b_d1) @ W_d2 + b_d2
// ---------------------------------------------------------------------------
__global__ __launch_bounds__(256)
void decoder_kernel(const float* __restrict__ new_x, const bf16_t* __restrict__ Wf,
                    const float* __restrict__ b_d1, const float* __restrict__ W_d2,
                    const float* __restrict__ b_d2, float* __restrict__ out, int nN) {
  __shared__ bf16_t ldsW[KK_D1 * 4 * 32 * 16];
  for (int i = threadIdx.x; i < KK_D1 * 4 * 32 * 16 / 8; i += blockDim.x)
    ((v8bf*)ldsW)[i] = ((const v8bf*)Wf)[i];
  __syncthreads();

  int lane = threadIdx.x & 31;
  int wave = threadIdx.x >> 5;
  int gw   = blockIdx.x * (blockDim.x >> 5) + wave;
  int nw   = gridDim.x * (blockDim.x >> 5);
  int row  = lane & 15;
  int koff = (lane >> 4) * 8;
  int coln = lane & 15;
  int mb   = (lane >> 4) * 8;
  int ntiles = nN >> 4;

  for (int tile = gw; tile < ntiles; tile += nw) {
    const float* xr = new_x + (size_t)(tile * 16 + row) * H;
    v8f acc[4] = {};
#pragma unroll
    for (int kk = 0; kk < KK_D1; ++kk) {
      int fA = kk * 32 + koff;
      v8bf lo, hi;
#pragma unroll
      for (int k = 0; k < 8; ++k) {
        lo[k] = (bf16_t)xr[fA + k];
        hi[k] = (bf16_t)xr[fA + 16 + k];
      }
      v16bf a = mk_frag(lo, hi);
#pragma unroll
      for (int c = 0; c < 4; ++c) {
        v16bf b = *(const v16bf*)(ldsW + ((((kk * 4 + c) * 32) + lane) << 4));
        acc[c] = WMMA_BF16(a, b, acc[c]);
      }
    }
    float p[8] = {0.f, 0.f, 0.f, 0.f, 0.f, 0.f, 0.f, 0.f};
#pragma unroll
    for (int c = 0; c < 4; ++c) {
      int n = c * 16 + coln;
      float bias = b_d1[n];
      float w2 = W_d2[n];
#pragma unroll
      for (int v = 0; v < 8; ++v) {
        float hval = acc[c][v] + bias;
        hval = hval > 0.f ? hval : 0.f;
        p[v] += hval * w2;
      }
    }
#pragma unroll
    for (int v = 0; v < 8; ++v) {
#pragma unroll
      for (int off = 1; off < 16; off <<= 1)
        p[v] += __shfl_xor(p[v], off, 32);
    }
    if (coln == 0) {
#pragma unroll
      for (int v = 0; v < 8; ++v)
        out[(size_t)tile * 16 + mb + v] = p[v] + b_d2[0];
    }
  }
}

// ---------------------------------------------------------------------------
extern "C" void kernel_launch(void* const* d_in, const int* in_sizes, int n_in,
                              void* d_out, int out_size, void* d_ws, size_t ws_size,
                              hipStream_t stream) {
  (void)in_sizes; (void)n_in; (void)out_size; (void)ws_size;

  const float* node_attr   = (const float*)d_in[0];
  const float* edge_attr   = (const float*)d_in[1];
  const float* global_attr = (const float*)d_in[2];
  const int*   edge_index  = (const int*)d_in[3];
  const float* W_ne = (const float*)d_in[5];
  const float* b_ne = (const float*)d_in[6];
  const float* W_ee = (const float*)d_in[7];
  const float* b_ee = (const float*)d_in[8];
  const float* W_ge = (const float*)d_in[9];
  const float* b_ge = (const float*)d_in[10];
  const float* W_eb = (const float*)d_in[11];
  const float* b_eb = (const float*)d_in[12];
  const float* W_nb = (const float*)d_in[13];
  const float* b_nb = (const float*)d_in[14];
  const float* W_gb = (const float*)d_in[15];
  const float* b_gb = (const float*)d_in[16];
  const float* W_d1 = (const float*)d_in[17];
  const float* b_d1 = (const float*)d_in[18];
  const float* W_d2 = (const float*)d_in[19];
  const float* b_d2 = (const float*)d_in[20];

  const int* senders   = edge_index;
  const int* receivers = edge_index + N_EDGES;

  // workspace layout
  char* ws = (char*)d_ws;
  size_t cur = 0;
  auto alloc = [&](size_t bytes) -> char* {
    char* p = ws + cur;
    cur += (bytes + 255) & ~(size_t)255;
    return p;
  };
  float*  enc_x = (float*)alloc((size_t)T_STEPS * N_NODES * H * sizeof(float));
  float*  new_e = (float*)alloc((size_t)N_EDGES * H * sizeof(float));
  float*  new_x = (float*)alloc((size_t)N_NODES * H * sizeof(float));
  bf16_t* cxb   = (bf16_t*)alloc((size_t)N_NODES * 128 * sizeof(bf16_t));
  float*  sent  = (float*)alloc((size_t)N_NODES * H * sizeof(float));
  float*  recv  = (float*)alloc((size_t)N_NODES * H * sizeof(float));
  bf16_t* Wf_ne = (bf16_t*)alloc((size_t)KK_NE * 4 * 32 * 16 * sizeof(bf16_t));
  bf16_t* Wf_eb = (bf16_t*)alloc((size_t)KK_EB * 4 * 32 * 16 * sizeof(bf16_t));
  bf16_t* Wf_nb = (bf16_t*)alloc((size_t)KK_NB * 4 * 32 * 16 * sizeof(bf16_t));
  bf16_t* Wf_d1 = (bf16_t*)alloc((size_t)KK_D1 * 4 * 32 * 16 * sizeof(bf16_t));
  bf16_t* gb    = (bf16_t*)alloc(H * sizeof(bf16_t));
  float*  h_g   = (float*)alloc(H * sizeof(float));
  float*  sumX  = (float*)alloc(H * sizeof(float));
  float*  sumE  = (float*)alloc(H * sizeof(float));

  // one-time weight repack (re-run every call; deterministic)
  repack_weights_kernel<<<1, 128, 0, stream>>>(W_ne, Wf_ne, KK_NE, F_IN);
  repack_weights_kernel<<<7, 256, 0, stream>>>(W_eb, Wf_eb, KK_EB, 448);
  repack_weights_kernel<<<5, 256, 0, stream>>>(W_nb, Wf_nb, KK_NB, 320);
  repack_weights_kernel<<<1, 256, 0, stream>>>(W_d1, Wf_d1, KK_D1, 64);

  encode_global_kernel<<<1, 64, 0, stream>>>(global_attr, W_ge, b_ge, h_g);
  encode_nodes_kernel<<<256, 256, 0, stream>>>(node_attr, Wf_ne, b_ne, enc_x,
                                               T_STEPS * N_NODES);

  float* out = (float*)d_out;
  for (int t = 0; t < T_STEPS; ++t) {
    const float* encx_t = enc_x + (size_t)t * N_NODES * H;
    const float* hx     = (t == 0) ? enc_x : new_x;

    prep_step_kernel<<<(N_NODES + 255) / 256, 256, 0, stream>>>(
        encx_t, hx, cxb, sent, recv, h_g, gb, sumX, sumE, N_NODES);

    edge_block_kernel<<<512, 256, 0, stream>>>(
        edge_attr + (size_t)t * N_EDGES, W_ee, b_ee, cxb, new_e,
        senders, receivers, Wf_eb, b_eb, gb, sent, recv, sumE,
        N_EDGES, (t == 0) ? 1 : 0);

    node_block_kernel<<<256, 256, 0, stream>>>(
        cxb, sent, recv, gb, Wf_nb, b_nb, new_x, sumX, N_NODES);

    global_block_kernel<<<1, 64, 0, stream>>>(sumX, sumE, h_g, W_gb, b_gb,
                                              N_NODES, N_EDGES);

    decoder_kernel<<<256, 256, 0, stream>>>(
        new_x, Wf_d1, b_d1, W_d2, b_d2, out + (size_t)t * N_NODES, N_NODES);
  }
}
